// LookupNCE_27822798144032
// MI455X (gfx1250) — matrix operand order repaired and checked by reading, not compiled, not run on hardware
//
#include <hip/hip_runtime.h>
#include <math.h>

// NCE loss, fused:  out[b] = softplus(-logit_true[b]) + sum_s softplus(inputs@w_samp^T + b_samp)
// B=8192, D=128, V=1e6, S=8192.
// GEMM path: split-bf16 (x = hi + lo), 3x V_WMMA_F32_16X16X32_BF16 per K-block
// (Ah*Bh + Ah*Bl + Al*Bh, fp32 accumulate) ~= fp32 precision at ~2.7x the fp32-WMMA rate.
// softplus + row-reduction fused into the GEMM epilogue (logits never materialized).

typedef __attribute__((ext_vector_type(16))) __bf16 v16bf;
typedef __attribute__((ext_vector_type(4)))  __bf16 v4bf;
typedef __attribute__((ext_vector_type(8)))  float  v8f;

#define BB 8192
#define DD 128
#define SS 8192
#define TB 128            // b-rows per block
#define TS 128            // s-cols per LDS tile
#define SCHUNK 1024       // s-cols per block
#define NSTILE (SCHUNK / TS)
#define LROW 136          // LDS row stride in bf16 (272 B): conflict-free half-wave b128 reads

__device__ __forceinline__ float softplus_f(float x) {
    float e = __expf(-fabsf(x));
    return fmaxf(x, 0.0f) + __logf(1.0f + e);
}

struct BF2 { __bf16 h, l; };
__device__ __forceinline__ BF2 split_bf16(float x) {
    BF2 r;
    r.h = (__bf16)x;
    r.l = (__bf16)(x - (float)r.h);
    return r;
}

__device__ __forceinline__ v8f bwmma(v16bf a, v16bf b, v8f c) {
    return __builtin_amdgcn_wmma_f32_16x16x32_bf16(
        /*neg_a=*/false, a, /*neg_b=*/false, b,
        /*c_mod=*/(short)0, c, /*reuse_a=*/false, /*reuse_b=*/false);
}

struct Frag { union { v16bf v; uint4 q[2]; }; };

// ---------------- kernel 1: true-class logits -> initialize out (exact fp32) ----
__global__ __launch_bounds__(256) void nce_true_kernel(
    const float* __restrict__ inputs, const float* __restrict__ weights,
    const float* __restrict__ biases, const int* __restrict__ true_ids,
    float* __restrict__ out)
{
    int b = blockIdx.x * 256 + threadIdx.x;
    const float4* ip = (const float4*)(inputs + (size_t)b * DD);
    int t = true_ids[b];
    const float4* wp = (const float4*)(weights + (size_t)t * DD);
    float acc = 0.0f;
#pragma unroll
    for (int i = 0; i < DD / 4; ++i) {
        float4 a = ip[i], w = wp[i];
        acc = fmaf(a.x, w.x, acc); acc = fmaf(a.y, w.y, acc);
        acc = fmaf(a.z, w.z, acc); acc = fmaf(a.w, w.w, acc);
    }
    out[b] = softplus_f(-(acc + biases[t]));
}

// ---------------- kernel 2: split inputs into bf16 hi/lo planes ----------------
__global__ __launch_bounds__(256) void nce_split_kernel(
    const float* __restrict__ x, __bf16* __restrict__ hi, __bf16* __restrict__ lo)
{
    int t = blockIdx.x * 256 + threadIdx.x;          // one float4 per thread
    float4 v = ((const float4*)x)[t];
    BF2 sx = split_bf16(v.x), sy = split_bf16(v.y);
    BF2 sz = split_bf16(v.z), sw = split_bf16(v.w);
    v4bf h, l;
    h.x = sx.h; h.y = sy.h; h.z = sz.h; h.w = sw.h;
    l.x = sx.l; l.y = sy.l; l.z = sz.l; l.w = sw.l;
    ((v4bf*)hi)[t] = h;
    ((v4bf*)lo)[t] = l;
}

// ---------------- kernel 3: gather sampled rows + split to bf16 hi/lo ----------
__global__ __launch_bounds__(256) void nce_gather_kernel(
    const float* __restrict__ weights, const float* __restrict__ biases,
    const int* __restrict__ sampled_ids, __bf16* __restrict__ w_hi,
    __bf16* __restrict__ w_lo, float* __restrict__ bsamp)
{
    int t = blockIdx.x * 256 + threadIdx.x;          // SS*32 threads, float4 each
    int s = t >> 5, j = t & 31;
    int sid = sampled_ids[s];
    float4 v = ((const float4*)(weights + (size_t)sid * DD))[j];
    BF2 sx = split_bf16(v.x), sy = split_bf16(v.y);
    BF2 sz = split_bf16(v.z), sw = split_bf16(v.w);
    v4bf h, l;
    h.x = sx.h; h.y = sy.h; h.z = sz.h; h.w = sw.h;
    l.x = sx.l; l.y = sy.l; l.z = sz.l; l.w = sw.l;
    ((v4bf*)(w_hi + (size_t)s * DD))[j] = h;
    ((v4bf*)(w_lo + (size_t)s * DD))[j] = l;
    if (j == 0) bsamp[s] = biases[sid];
}

// ---------------- GEMM helpers -------------------------------------------------
// cooperative 128x128 bf16 tile load: global (row stride 128) -> LDS (row stride 136)
__device__ __forceinline__ void load_tile(const __bf16* __restrict__ g, __bf16* lds)
{
    const uint4* gq = (const uint4*)g;   // 16 uint4 per row
    uint4* lq = (uint4*)lds;             // 17 uint4 per row (pad)
#pragma unroll
    for (int i = 0; i < 8; ++i) {
        int idx = (int)threadIdx.x + i * 256;   // 0..2047
        int r = idx >> 4, c = idx & 15;
        lq[r * 17 + c] = gq[r * 16 + c];
    }
}

// A fragment (16x32 bf16): lane<16 K={0..7,16..23}, lane>=16 K={8..15,24..31}
__device__ __forceinline__ Frag fragA(const __bf16* lds, int row, int kb, int hi)
{
    const char* base = (const char*)(lds + row * LROW) + kb * 64 + hi * 16;
    Frag f;
    f.q[0] = *(const uint4*)(base);
    f.q[1] = *(const uint4*)(base + 32);
    return f;
}

// B fragment (32x16 bf16): lane<16 K={0..7,16..23}, lane>=16 K={8..15,24..31}
__device__ __forceinline__ Frag fragB(const __bf16* lds, int row, int kb, int hi)
{
    const char* base = (const char*)(lds + row * LROW) + kb * 64 + hi * 16;
    Frag f;
    f.q[0] = *(const uint4*)(base);
    f.q[1] = *(const uint4*)(base + 32);
    return f;
}

// ---------------- kernel 4: fused GEMM + bias + softplus + row-reduce ----------
__global__ __launch_bounds__(256) void nce_gemm_kernel(
    const __bf16* __restrict__ a_hi, const __bf16* __restrict__ a_lo,
    const __bf16* __restrict__ w_hi, const __bf16* __restrict__ w_lo,
    const float* __restrict__ bsamp, float* __restrict__ out)
{
    __shared__ __bf16 ldsAh[TB * LROW], ldsAl[TB * LROW];   // 34 KB each
    __shared__ __bf16 ldsBh[TS * LROW], ldsBl[TS * LROW];   // total 136 KB

    const int lane = threadIdx.x & 31;
    const int wave = threadIdx.x >> 5;
    const int ln16 = lane & 15;
    const int hi   = lane >> 4;
    const int mg   = wave >> 1;          // 0..3: 32-row group
    const int cg   = wave & 1;           // 0..1: 64-col group
    const int b0 = blockIdx.x * TB;
    const int s0 = blockIdx.y * SCHUNK;

    load_tile(a_hi + (size_t)b0 * DD, ldsAh);
    load_tile(a_lo + (size_t)b0 * DD, ldsAl);

    float rs[2][8];
#pragma unroll
    for (int tm = 0; tm < 2; ++tm)
#pragma unroll
        for (int r = 0; r < 8; ++r) rs[tm][r] = 0.0f;

#pragma unroll 1
    for (int st = 0; st < NSTILE; ++st) {
        const int sbase = s0 + st * TS;

        __syncthreads();                 // previous tile fully consumed
        load_tile(w_hi + (size_t)sbase * DD, ldsBh);
        load_tile(w_lo + (size_t)sbase * DD, ldsBl);
        if (st + 1 < NSTILE) {           // prefetch next tile into caches
            __builtin_prefetch((const char*)(w_hi + (size_t)(sbase + TS) * DD) + threadIdx.x * 128, 0, 0);
            __builtin_prefetch((const char*)(w_lo + (size_t)(sbase + TS) * DD) + threadIdx.x * 128, 0, 0);
        }
        __syncthreads();

        v8f acc[2][4] = {};              // 32(b) x 64(s) per wave

#pragma unroll
        for (int kb = 0; kb < 4; ++kb) {
            Frag ah0 = fragA(ldsAh, mg * 32 + ln16,      kb, hi);
            Frag ah1 = fragA(ldsAh, mg * 32 + 16 + ln16, kb, hi);
            Frag al0 = fragA(ldsAl, mg * 32 + ln16,      kb, hi);
            Frag al1 = fragA(ldsAl, mg * 32 + 16 + ln16, kb, hi);
#pragma unroll
            for (int tn = 0; tn < 4; ++tn) {
                const int brow = cg * 64 + tn * 16 + ln16;
                Frag bh = fragB(ldsBh, brow, kb, hi);
                Frag bl = fragB(ldsBl, brow, kb, hi);
                acc[0][tn] = bwmma(ah0.v, bh.v, acc[0][tn]);
                acc[1][tn] = bwmma(ah1.v, bh.v, acc[1][tn]);
                acc[0][tn] = bwmma(ah0.v, bl.v, acc[0][tn]);
                acc[1][tn] = bwmma(ah1.v, bl.v, acc[1][tn]);
                acc[0][tn] = bwmma(al0.v, bh.v, acc[0][tn]);
                acc[1][tn] = bwmma(al1.v, bh.v, acc[1][tn]);
            }
        }

        // epilogue: C layout -> VGPR r holds M=r (lanes<16) / M=8+r (lanes>=16), N=lane&15
#pragma unroll
        for (int tn = 0; tn < 4; ++tn) {
            float bias = bsamp[sbase + cg * 64 + tn * 16 + ln16];
#pragma unroll
            for (int tm = 0; tm < 2; ++tm)
#pragma unroll
                for (int r = 0; r < 8; ++r)
                    rs[tm][r] += softplus_f(acc[tm][tn][r] + bias);
        }
    }

    // reduce across the 16 lanes of each half (masks < 16 stay within half)
#pragma unroll
    for (int tm = 0; tm < 2; ++tm)
#pragma unroll
        for (int r = 0; r < 8; ++r) {
            float v = rs[tm][r];
            v += __shfl_xor(v, 1, 32);
            v += __shfl_xor(v, 2, 32);
            v += __shfl_xor(v, 4, 32);
            v += __shfl_xor(v, 8, 32);
            if (ln16 == 0)
                atomicAdd(&out[b0 + mg * 32 + tm * 16 + 8 * hi + r], v);
        }
}

// ---------------------------------- launcher -----------------------------------
extern "C" void kernel_launch(void* const* d_in, const int* in_sizes, int n_in,
                              void* d_out, int out_size, void* d_ws, size_t ws_size,
                              hipStream_t stream)
{
    const float* inputs      = (const float*)d_in[0];   // [B, D]
    const float* weights     = (const float*)d_in[1];   // [V, D]
    const float* biases      = (const float*)d_in[2];   // [V]
    const int*   true_ids    = (const int*)d_in[3];     // [B]
    const int*   sampled_ids = (const int*)d_in[4];     // [S]
    float* out = (float*)d_out;                         // [B]

    // workspace layout (all 16B aligned): 4 bf16 planes of 2 MB + bias array
    __bf16* w_hi = (__bf16*)d_ws;
    __bf16* w_lo = w_hi + (size_t)SS * DD;
    __bf16* a_hi = w_lo + (size_t)SS * DD;
    __bf16* a_lo = a_hi + (size_t)BB * DD;
    float*  bsamp = (float*)(a_lo + (size_t)BB * DD);

    nce_true_kernel<<<BB / 256, 256, 0, stream>>>(inputs, weights, biases, true_ids, out);
    nce_split_kernel<<<(BB * DD / 4) / 256, 256, 0, stream>>>(inputs, a_hi, a_lo);
    nce_gather_kernel<<<(SS * 32) / 256, 256, 0, stream>>>(weights, biases, sampled_ids,
                                                           w_hi, w_lo, bsamp);
    dim3 grid(BB / TB, SS / SCHUNK);                    // 64 x 8 = 512 blocks
    nce_gemm_kernel<<<grid, 256, 0, stream>>>(a_hi, a_lo, w_hi, w_lo, bsamp, out);
}
